// Head_34479997452852
// MI455X (gfx1250) — matrix operand order repaired and checked by reading, not compiled, not run on hardware
//
#include <hip/hip_runtime.h>
#include <hip/hip_bf16.h>

// Problem constants (match reference)
constexpr int B = 4;
constexpr int T = 4096;
constexpr int C = 768;
constexpr int H = 64;

typedef __attribute__((ext_vector_type(16))) _Float16 v16h;
typedef __attribute__((ext_vector_type(8)))  _Float16 v8h;
typedef __attribute__((ext_vector_type(8)))  float    v8f;
typedef __attribute__((ext_vector_type(4)))  unsigned u32x4;
typedef __attribute__((ext_vector_type(8)))  unsigned u32x8;

// -----------------------------------------------------------------------------
// gfx1250 data movers
// -----------------------------------------------------------------------------

// Per-lane async copy of 16 bytes global -> LDS (tracked by ASYNCcnt).
__device__ __forceinline__ void async_copy_b128(unsigned lds_off, const void* g) {
  unsigned long long ga = (unsigned long long)(uintptr_t)g;
  asm volatile("global_load_async_to_lds_b128 %0, %1, off"
               :: "v"(lds_off), "v"(ga) : "memory");
}

__device__ __forceinline__ void wait_asynccnt0() {
  asm volatile("s_wait_asynccnt 0x0" ::: "memory");
}

// TDM 2-D tile load: tile_d1 rows x tile_d0 elements (f16), row stride
// stride_elems, with LDS padding of 4 DWORDs after every 32 DWORDs
// (i.e. 64-half rows land at a 72-half padded stride in LDS).
// Descriptor layout per CDNA5 ISA.
__device__ __forceinline__ void tdm_load_2d_f16(unsigned lds_off, const void* gaddr,
                                                unsigned tile_d0, unsigned tile_d1,
                                                unsigned stride_elems,
                                                unsigned tensor_d1) {
  unsigned long long ga = (unsigned long long)(uintptr_t)gaddr;
  u32x4 g0;
  g0[0] = 1u;                                   // count=1, user descriptor
  g0[1] = lds_off;                              // lds_addr (bytes)
  g0[2] = (unsigned)ga;                         // global_addr[31:0]
  g0[3] = ((unsigned)(ga >> 32) & 0x01ffffffu) | 0x80000000u; // addr[56:32] | type=2
  u32x8 g1;
  // data_size=1 (2B), pad_enable, pad_interval=4 (32 DWORDs), pad_amount=3 (4 DWORDs)
  g1[0] = (1u << 16) | (1u << 20) | (4u << 22) | (3u << 25);
  g1[1] = (tile_d0 & 0xffffu) << 16;            // tensor_dim0[15:0] (= row length)
  g1[2] = (tensor_d1 & 0xffffu) << 16;          // tensor_dim0[31:16]=0 | tensor_dim1[15:0]
  g1[3] = ((tensor_d1 >> 16) & 0xffffu) | ((tile_d0 & 0xffffu) << 16); // dim1 hi | tile_dim0
  g1[4] = tile_d1 & 0xffffu;                    // tile_dim1 | tile_dim2=0
  g1[5] = stride_elems;                         // tensor_dim0_stride[31:0]
  g1[6] = 0u;                                   // stride hi | tensor_dim1_stride lo
  g1[7] = 0u;
  asm volatile("tensor_load_to_lds %0, %1" :: "s"(g0), "s"(g1) : "memory");
}

// -----------------------------------------------------------------------------
// Kernel 0: W [C][H] f32  ->  W^T [H][C] f16  (makes WMMA B-fragments contiguous)
// -----------------------------------------------------------------------------
__global__ void head_wtrans(const float* __restrict__ W, _Float16* __restrict__ WT) {
  int i = blockIdx.x * blockDim.x + threadIdx.x;   // over H*C
  if (i >= H * C) return;
  int h = i / C, c = i % C;
  WT[i] = (_Float16)W[c * H + h];
}

// -----------------------------------------------------------------------------
// Kernel 1: QKV projection with WMMA f16 -> f32 acc, store q/k/v as f16.
// One wave computes 16 rows x 64 cols for all three matrices.
// Block = 128 threads (4 waves), grid = B*T/64.
// -----------------------------------------------------------------------------
__global__ __launch_bounds__(128)
void head_qkv_proj(const float* __restrict__ x,
                   const _Float16* __restrict__ WqT,
                   const _Float16* __restrict__ WkT,
                   const _Float16* __restrict__ WvT,
                   _Float16* __restrict__ qh,
                   _Float16* __restrict__ kh,
                   _Float16* __restrict__ vh) {
  const int lane  = threadIdx.x & 31;
  const int wid   = threadIdx.x >> 5;
  const int tbase = blockIdx.x * 64 + wid * 16;     // flattened row over B*T
  const int m     = lane & 15;
  const int khalf = lane >> 4;

  const _Float16* WT[3] = {WqT, WkT, WvT};
  v8f zero = {};
  v8f acc[3][4];
#pragma unroll
  for (int mat = 0; mat < 3; ++mat)
#pragma unroll
    for (int j = 0; j < 4; ++j) acc[mat][j] = zero;

  for (int kc = 0; kc < C / 32; ++kc) {
    const int cbase = kc * 32;
    // A fragment: contiguous K order. elem i -> K=16*khalf+i
    v16h a;
    const float* xr = x + (size_t)(tbase + m) * C + cbase + 16 * khalf;
#pragma unroll
    for (int i = 0; i < 8; ++i) {
      a[i]     = (_Float16)xr[i];
      a[8 + i] = (_Float16)xr[8 + i];
    }
#pragma unroll
    for (int mat = 0; mat < 3; ++mat) {
#pragma unroll
      for (int j = 0; j < 4; ++j) {
        // B fragment: lane -> col N=m (via W^T row), elem e -> K = 16*khalf + e (contiguous)
        const _Float16* wp = WT[mat] + (size_t)(j * 16 + m) * C + cbase + 16 * khalf;
        v16h b = *(const v16h*)wp;   // 32B contiguous, 32B aligned
        acc[mat][j] = __builtin_amdgcn_wmma_f32_16x16x32_f16(
            false, a, false, b, (short)0, acc[mat][j], false, false);
      }
    }
  }

  _Float16* outp[3] = {qh, kh, vh};
#pragma unroll
  for (int mat = 0; mat < 3; ++mat)
#pragma unroll
    for (int j = 0; j < 4; ++j)
#pragma unroll
      for (int r = 0; r < 8; ++r) {
        // D layout: M = r + 8*khalf, N = m
        size_t t = (size_t)tbase + r + 8 * khalf;
        outp[mat][t * H + j * 16 + m] = (_Float16)acc[mat][j][r];
      }
}

// -----------------------------------------------------------------------------
// Kernel 2: fused causal flash attention.
// Block = 128 threads (4 waves) handles 64 query rows of one batch element.
// Q tile staged by TDM (tensor_load_to_lds, TENSORcnt); K tiles staged by
// global_load_async_to_lds_b128 (ASYNCcnt); V transposed through VGPRs.
// -----------------------------------------------------------------------------
constexpr int QSTR = 72;  // padded f16 stride produced by TDM pad (64 + 8)
constexpr int KSTR = 72;  // padded f16 stride for K tile rows
constexpr int VSTR = 40;  // padded f16 stride for V^T rows
constexpr int PSTR = 40;  // padded f16 stride for per-wave P tile

__global__ __launch_bounds__(128)
void head_attn(const _Float16* __restrict__ qh,
               const _Float16* __restrict__ kh,
               const _Float16* __restrict__ vh,
               float* __restrict__ out) {
  __shared__ alignas(16) _Float16 qsh[64 * QSTR];        // Q tile (TDM, padded)
  __shared__ alignas(16) _Float16 ksh[32 * KSTR];        // K tile, row-major [s][h]
  __shared__ alignas(16) _Float16 vsh[64 * VSTR];        // V tile transposed [h][s]
  __shared__ alignas(16) _Float16 psh[4 * 16 * PSTR];    // per-wave P tiles [16][32]

  const int lane  = threadIdx.x & 31;
  const int wid   = threadIdx.x >> 5;
  const int m     = lane & 15;
  const int khalf = lane >> 4;

  const int ntq = T / 64;
  const int b   = blockIdx.x / ntq;
  const int qt  = blockIdx.x % ntq;
  const size_t rowbase  = (size_t)b * T + qt * 64;  // flattened first row of q tile
  const int    qrowtile = qt * 64;                  // position within T

  // ---- stage Q tile (64x64 f16, contiguous 8KB) into LDS via TDM ----
  if (wid == 0) {
    tdm_load_2d_f16((unsigned)(uintptr_t)&qsh[0], qh + rowbase * H,
                    /*tile_d0=*/H, /*tile_d1=*/64, /*stride=*/H, /*tensor_d1=*/64);
    __builtin_amdgcn_s_wait_tensorcnt(0);
  }
  __syncthreads();

  // ---- Q fragments (K=0..31 and K=32..63), A layout, from padded LDS ----
  v16h qfrag[2];
  {
    const _Float16* qr = &qsh[(wid * 16 + m) * QSTR];
#pragma unroll
    for (int kc = 0; kc < 2; ++kc) {
      const _Float16* p0 = qr + kc * 32 + 8 * khalf;
      v8h lo = *(const v8h*)p0;
      v8h hi = *(const v8h*)(p0 + 16);
#pragma unroll
      for (int i = 0; i < 8; ++i) {
        qfrag[kc][i]     = lo[i];
        qfrag[kc][8 + i] = hi[i];
      }
    }
  }

  v8f zero = {};
  v8f o[4];
#pragma unroll
  for (int jo = 0; jo < 4; ++jo) o[jo] = zero;
  float mrow[8], lrow[8];
#pragma unroll
  for (int r = 0; r < 8; ++r) { mrow[r] = -3.0e38f; lrow[r] = 0.0f; }

  const float scale = 0.03608439182435161f;  // C**-0.5 (reference scales by embed dim)
  const int s_end = qrowtile + 64;           // causal bound for this q tile

  for (int s0 = 0; s0 < s_end; s0 += 32) {
    // ---- K tile: per-lane async copy global -> LDS (padded rows) ----
    {
      const int tid  = threadIdx.x;
      const int row  = tid >> 2;       // 0..31
      const int chnk = tid & 3;        // 0..3  (16 halfs each)
      const _Float16* gk = kh + ((size_t)b * T + s0 + row) * H + chnk * 16;
      unsigned loff = (unsigned)(uintptr_t)&ksh[row * KSTR + chnk * 16];
      async_copy_b128(loff, gk);
      async_copy_b128(loff + 16u, gk + 8);

      // ---- V tile: transpose through VGPRs into vsh (overlaps async K) ----
      const _Float16* gv = vh + ((size_t)b * T + s0 + row) * H + chnk * 16;
      v8h u0 = *(const v8h*)(gv);
      v8h u1 = *(const v8h*)(gv + 8);
#pragma unroll
      for (int i = 0; i < 8; ++i) {
        vsh[(chnk * 16 + i) * VSTR + row]     = u0[i];
        vsh[(chnk * 16 + 8 + i) * VSTR + row] = u1[i];
      }
      if (s0 + 32 < s_end) {  // prefetch next KV tile (global_prefetch_b8)
        __builtin_prefetch(gk + 32 * H, 0, 3);
        __builtin_prefetch(gv + 32 * H, 0, 3);
      }
    }
    wait_asynccnt0();
    __syncthreads();

    // ---- S = Q K^T : two 16-col blocks, K-dim 64 = 2 x WMMA ----
    v8f s[2];
#pragma unroll
    for (int j = 0; j < 2; ++j) {
      v8f accs = zero;
#pragma unroll
      for (int kc = 0; kc < 2; ++kc) {
        const _Float16* kp = &ksh[(j * 16 + m) * KSTR + kc * 32 + 8 * khalf];
        v8h lo = *(const v8h*)kp;
        v8h hi = *(const v8h*)(kp + 16);
        v16h bfr;
#pragma unroll
        for (int i = 0; i < 8; ++i) { bfr[i] = lo[i]; bfr[8 + i] = hi[i]; }
        accs = __builtin_amdgcn_wmma_f32_16x16x32_f16(
            false, qfrag[kc], false, bfr, (short)0, accs, false, false);
      }
      s[j] = accs;
    }

    // ---- scale + causal mask (D layout: row M = r + 8*khalf, col N = m) ----
    const int qr0 = qrowtile + wid * 16 + 8 * khalf;
#pragma unroll
    for (int j = 0; j < 2; ++j) {
      const int col = s0 + j * 16 + m;
#pragma unroll
      for (int r = 0; r < 8; ++r) {
        float v = s[j][r] * scale;
        s[j][r] = (col <= qr0 + r) ? v : -3.0e38f;
      }
    }

    // ---- online softmax (rows live on 16-lane halves; xor<=8 stays in-half) ----
    float alpha[8];
#pragma unroll
    for (int r = 0; r < 8; ++r) {
      float tm = fmaxf(s[0][r], s[1][r]);
#pragma unroll
      for (int off = 8; off >= 1; off >>= 1)
        tm = fmaxf(tm, __shfl_xor(tm, off, 32));
      float mn = fmaxf(mrow[r], tm);
      alpha[r] = __expf(mrow[r] - mn);
      mrow[r]  = mn;
      float p0 = __expf(s[0][r] - mn);
      float p1 = __expf(s[1][r] - mn);
      s[0][r] = p0;
      s[1][r] = p1;
      float rs = p0 + p1;
#pragma unroll
      for (int off = 8; off >= 1; off >>= 1)
        rs += __shfl_xor(rs, off, 32);
      lrow[r] = lrow[r] * alpha[r] + rs;
    }
#pragma unroll
    for (int jo = 0; jo < 4; ++jo)
#pragma unroll
      for (int r = 0; r < 8; ++r) o[jo][r] *= alpha[r];

    // ---- P (D layout) -> per-wave LDS -> A layout fragment ----
    _Float16* pw = &psh[wid * 16 * PSTR];
#pragma unroll
    for (int j = 0; j < 2; ++j)
#pragma unroll
      for (int r = 0; r < 8; ++r)
        pw[(r + 8 * khalf) * PSTR + j * 16 + m] = (_Float16)s[j][r];

    v16h pa;
    {
      const _Float16* pr = &pw[m * PSTR];
#pragma unroll
      for (int i = 0; i < 8; ++i) {
        pa[i]     = pr[8 * khalf + i];
        pa[8 + i] = pr[16 + 8 * khalf + i];
      }
    }

    // ---- O += P @ V  (B fragment contiguous from transposed V tile) ----
#pragma unroll
    for (int jo = 0; jo < 4; ++jo) {
      const _Float16* vp = &vsh[(jo * 16 + m) * VSTR + 8 * khalf];
      v8h lo = *(const v8h*)vp;
      v8h hi = *(const v8h*)(vp + 16);
      v16h bfr;
#pragma unroll
      for (int i = 0; i < 8; ++i) { bfr[i] = lo[i]; bfr[8 + i] = hi[i]; }
      o[jo] = __builtin_amdgcn_wmma_f32_16x16x32_f16(
          false, pa, false, bfr, (short)0, o[jo], false, false);
    }
    __syncthreads();  // protect ksh/vsh before next tile load
  }

  // ---- epilogue: normalize and store fp32 output [B,T,H] ----
#pragma unroll
  for (int jo = 0; jo < 4; ++jo)
#pragma unroll
    for (int r = 0; r < 8; ++r) {
      size_t t = rowbase + wid * 16 + r + 8 * khalf;
      out[t * H + jo * 16 + m] = o[jo][r] / lrow[r];
    }
}

// -----------------------------------------------------------------------------
// Launcher
// -----------------------------------------------------------------------------
extern "C" void kernel_launch(void* const* d_in, const int* in_sizes, int n_in,
                              void* d_out, int out_size, void* d_ws, size_t ws_size,
                              hipStream_t stream) {
  const float* x  = (const float*)d_in[0];
  const float* Wq = (const float*)d_in[1];
  const float* Wk = (const float*)d_in[2];
  const float* Wv = (const float*)d_in[3];
  float* out = (float*)d_out;

  // workspace layout (f16): q | k | v | WqT | WkT | WvT   (~6.6 MB total)
  _Float16* ws  = (_Float16*)d_ws;
  _Float16* qhp = ws;
  _Float16* khp = qhp + (size_t)B * T * H;
  _Float16* vhp = khp + (size_t)B * T * H;
  _Float16* WqT = vhp + (size_t)B * T * H;
  _Float16* WkT = WqT + (size_t)H * C;
  _Float16* WvT = WkT + (size_t)H * C;

  const int nwt = (H * C + 255) / 256;
  head_wtrans<<<dim3(nwt), dim3(256), 0, stream>>>(Wq, WqT);
  head_wtrans<<<dim3(nwt), dim3(256), 0, stream>>>(Wk, WkT);
  head_wtrans<<<dim3(nwt), dim3(256), 0, stream>>>(Wv, WvT);

  head_qkv_proj<<<dim3(B * T / 64), dim3(128), 0, stream>>>(
      x, WqT, WkT, WvT, qhp, khp, vhp);

  head_attn<<<dim3(B * (T / 64)), dim3(128), 0, stream>>>(qhp, khp, vhp, out);
}
